// SelfAttn2D_48515950575815
// MI455X (gfx1250) — matrix operand order repaired and checked by reading, not compiled, not run on hardware
//
#include <hip/hip_runtime.h>

typedef __bf16 bf16_t;
typedef __attribute__((ext_vector_type(16))) __bf16 v16bf;
typedef __attribute__((ext_vector_type(8)))  __bf16 v8bf;
typedef __attribute__((ext_vector_type(8)))  float  v8f;
typedef __attribute__((ext_vector_type(4)))  unsigned int u32x4;
typedef __attribute__((ext_vector_type(8)))  unsigned int u32x8;

#define B_DIM 8
#define C_DIM 256
#define CQK   32
#define N_DIM 4096            // 64*64
#define KV_ELTS 9216          // bf16 elements per (b,jt32) block: 2*512 (K) + 16*512 (V)
#define KV_BYTES 18432

// ---------------------------------------------------------------------------
// Kernel 1: q/k/v projections (1x1 convs), written as bf16 WMMA fragments.
//   qB[b][it][lane][e] : B-frag of Q [32ch x 16q], ch = (lane>=16?16:0)+e
//   kv[b][jt32]        : 18KB block = 2x K A-frag tiles (16key x 32ch)
//                        followed by 16x V A-frag tiles (16c x 32j)
//   A-frag map: Kmap(h,e) = h*16 + e  (contiguous, same order as the Q/P B-frags)
// ---------------------------------------------------------------------------
__global__ __launch_bounds__(256) void sa_proj_kernel(
    const float* __restrict__ x,
    const float* __restrict__ Wq, const float* __restrict__ bq,
    const float* __restrict__ Wk, const float* __restrict__ bk,
    const float* __restrict__ Wv, const float* __restrict__ bv,
    bf16_t* __restrict__ qB, bf16_t* __restrict__ kv)
{
    __shared__ float xs[C_DIM][33];           // 32 cols + pad
    const int b  = blockIdx.x >> 7;           // 128 column-tiles per batch
    const int nt = blockIdx.x & 127;
    const int n0 = nt * 32;
    const int t  = threadIdx.x;

    // stage x[b][c][n0:n0+32] into LDS (thread t loads row c=t)
    {
        const float* src = x + ((size_t)b * C_DIM + t) * N_DIM + n0;
        #pragma unroll
        for (int j = 0; j < 32; j += 4) {
            float4 v4 = *(const float4*)(src + j);
            xs[t][j+0] = v4.x; xs[t][j+1] = v4.y;
            xs[t][j+2] = v4.z; xs[t][j+3] = v4.w;
        }
    }
    __syncthreads();

    const int n_l  = t & 31;          // local column
    const int og   = t >> 5;          // wave-uniform output-group 0..7
    const int n    = n0 + n_l;        // global column (query/key index)
    const int it   = n >> 4;          // 16-wide tile index (0..255)
    const int jt32 = n >> 5;          // 32-wide tile index (0..127)
    bf16_t* kvb = kv + (size_t)(b * 128 + jt32) * KV_ELTS;

    // ---- q and k (32 output channels) ----
    for (int oc = og; oc < CQK; oc += 8) {
        float accq = bq[oc], acck = bk[oc];
        const float* wq = Wq + oc * C_DIM;
        const float* wk = Wk + oc * C_DIM;
        #pragma unroll 8
        for (int c = 0; c < C_DIM; ++c) {
            float xv = xs[c][n_l];
            accq = fmaf(wq[c], xv, accq);
            acck = fmaf(wk[c], xv, acck);
        }
        {   // q -> B fragment: ch = half*16 + e
            int lane = (n & 15) + ((oc >= 16) ? 16 : 0);
            int e    = oc & 15;
            qB[(((size_t)b * 256 + it) * 32 + lane) * 16 + e] = (bf16_t)accq;
        }
        {   // k -> A fragment inside kv block (tile tt = (n>>4)&1)
            int tt   = (n >> 4) & 1;
            int half = (oc >> 4) & 1;
            int e    = oc & 15;
            int lane = (n & 15) + half * 16;
            kvb[tt * 512 + lane * 16 + e] = (bf16_t)acck;
        }
    }

    // ---- v (256 output channels) ----
    for (int oc = og; oc < C_DIM; oc += 8) {
        float accv = bv[oc];
        const float* wv = Wv + oc * C_DIM;
        #pragma unroll 8
        for (int c = 0; c < C_DIM; ++c)
            accv = fmaf(wv[c], xs[c][n_l], accv);
        int ct   = oc >> 4;
        int r    = n & 31;                     // j within 32-tile
        int half = (r >> 4) & 1;
        int e    = r & 15;
        int lane = (oc & 15) + half * 16;
        kvb[1024 + (ct * 32 + lane) * 16 + e] = (bf16_t)accv;
    }
}

// ---------------------------------------------------------------------------
// TDM: 1-D tile load (KV_BYTES contiguous) global -> LDS.
// D# group0: count=1 | lds_addr | global_addr(57b) | type=2
// D# group1: data_size=3 (8B units), tensor_dim0=tile_dim0=2304, dim1=1
// ---------------------------------------------------------------------------
__device__ __forceinline__ void tdm_load_kv(unsigned lds_addr, const void* gptr) {
    unsigned long long ga = (unsigned long long)(uintptr_t)gptr;
    u32x4 g0 = { 1u,                                    // count=1, no gather
                 lds_addr,
                 (unsigned)(ga & 0xffffffffu),
                 ((unsigned)(ga >> 32) & 0x01ffffffu) | 0x80000000u }; // type=2
    u32x8 g1 = { 0x00030000u,                           // data_size=3 (8B)
                 (2304u & 0xffffu) << 16,               // tensor_dim0 lo
                 0x00010000u,                           // tensor_dim1 = 1
                 2304u << 16,                           // tile_dim0 = 2304
                 0u,                                    // tile_dim1/2 = 0
                 2304u,                                 // tensor_dim0_stride
                 0u, 0u };
    asm volatile("tensor_load_to_lds %0, %1" :: "s"(g0), "s"(g1) : "memory");
}

// ---------------------------------------------------------------------------
// Kernel 2: flash attention. 4 waves/block share one batch; wave w owns query
// tile it0+w. Wave 0 (scalar-branched) triple-buffers the 18KB K+V fragment
// block per 32-key step via the Tensor Data Mover.
//   energy^T tile = wmma(kA, qB)   -> C: M=key, N=query
//   P (bf16) staged in LDS [i][j]  -> B-frag of P^T
//   out^T tile += wmma(vA, P^T)    -> C: M=channel, N=query
// ---------------------------------------------------------------------------
__global__ __launch_bounds__(128) void sa_attn_kernel(
    const bf16_t* __restrict__ qB, const bf16_t* __restrict__ kv,
    const float* __restrict__ x, const float* __restrict__ gamma,
    float* __restrict__ out)
{
    __shared__ __align__(32) bf16_t stage[3 * KV_ELTS];   // 54KB K+V staging
    __shared__ __align__(32) bf16_t Pbuf[4][16][32];      // per-wave P tiles

    const int lane = threadIdx.x & 31;
    const int w    = threadIdx.x >> 5;
    // wave id as a true SGPR value -> scalar branch around TDM/waits,
    // so only wave 0 issues DMA (TDM ignores EXEC; SOPP waits ignore EXEC).
    const int sw   = __builtin_amdgcn_readfirstlane(threadIdx.x) >> 5;
    const int blk  = blockIdx.x;           // 0..511
    const int b    = blk >> 6;             // 64 blocks per batch
    const int it   = (blk & 63) * 4 + w;   // query tile 0..255
    const int half = lane >> 4;
    const int lm   = lane & 15;

    const bf16_t* kvb = kv + (size_t)b * 128 * KV_ELTS;
    const unsigned stage_base = (unsigned)(uintptr_t)&stage[0];

    // Q B-fragment for this query tile (lives for the whole loop)
    const v16bf qf = *(const v16bf*)(qB + (((size_t)b * 256 + it) * 32 + lane) * 16);

    v8f acc[16];
    #pragma unroll
    for (int ct = 0; ct < 16; ++ct) { v8f z = {}; acc[ct] = z; }

    float m = -3.0e38f, l = 0.0f;
    const v8f zero = {};

    if (sw == 0) {                                 // prime buffers 0 and 1
        tdm_load_kv(stage_base, kvb);
        tdm_load_kv(stage_base + KV_BYTES, kvb + KV_ELTS);
    }

    int cur = 0;                                   // jt % 3 (rotating)
    for (int jt = 0; jt < 128; ++jt) {   // 32 keys per step
        if (sw == 0) {
            // in-order TDM completion: <=1 outstanding => load jt done
            if (jt < 127) __builtin_amdgcn_s_wait_tensorcnt(1);
            else          __builtin_amdgcn_s_wait_tensorcnt(0);
        }
        __syncthreads();                           // buffer cur ready for all;
                                                   // readers of buffer (cur+2)%3 retired
        if (sw == 0 && jt + 2 < 128) {             // 2-deep prefetch
            int nb = cur + 2; if (nb >= 3) nb -= 3;
            tdm_load_kv(stage_base + nb * KV_BYTES,
                        kvb + (size_t)(jt + 2) * KV_ELTS);
        }

        const bf16_t* bufp = &stage[cur * KV_ELTS];
        v16bf kf0 = *(const v16bf*)(bufp + lane * 16);
        v16bf kf1 = *(const v16bf*)(bufp + 512 + lane * 16);

        // energy^T tiles: lane holds j = jt*32 + {half*8+r, 16+half*8+r}, i = lm
        v8f e0 = __builtin_amdgcn_wmma_f32_16x16x32_bf16(false, kf0, false, qf,
                                                         (short)0, zero, false, false);
        v8f e1 = __builtin_amdgcn_wmma_f32_16x16x32_bf16(false, kf1, false, qf,
                                                         (short)0, zero, false, false);

        // online softmax over j for query i = lm (shared by lane and lane^16)
        float mloc = -3.0e38f;
        #pragma unroll
        for (int r = 0; r < 8; ++r) {
            mloc = fmaxf(mloc, e0[r]);
            mloc = fmaxf(mloc, e1[r]);
        }
        mloc = fmaxf(mloc, __shfl_xor(mloc, 16, 32));
        float mnew = fmaxf(m, mloc);
        float corr = __expf(m - mnew);
        m = mnew;

        float rsum = 0.0f;
        float p0[8], p1[8];
        #pragma unroll
        for (int r = 0; r < 8; ++r) {
            p0[r] = __expf(e0[r] - mnew);
            p1[r] = __expf(e1[r] - mnew);
            rsum += p0[r] + p1[r];
        }
        rsum += __shfl_xor(rsum, 16, 32);
        l = l * corr + rsum;

        // rescale running output accumulators
        #pragma unroll
        for (int ct = 0; ct < 16; ++ct)
            #pragma unroll
            for (int r = 0; r < 8; ++r)
                acc[ct][r] *= corr;

        // P -> LDS in [i][j] order (contiguous 8x bf16 per lane per sub-tile)
        v8bf s0, s1;
        #pragma unroll
        for (int r = 0; r < 8; ++r) { s0[r] = (bf16_t)p0[r]; s1[r] = (bf16_t)p1[r]; }
        *(v8bf*)&Pbuf[w][lm][half * 8]      = s0;
        *(v8bf*)&Pbuf[w][lm][16 + half * 8] = s1;
        asm volatile("s_wait_dscnt 0" ::: "memory");

        // P^T B-fragment: lane -> i = lm, j = half*16 + e  (16 contiguous bf16)
        v16bf pf = *(const v16bf*)&Pbuf[w][lm][half * 16];

        // out^T += V * P^T  over all 16 channel tiles (V frags from LDS stage)
        #pragma unroll
        for (int ct = 0; ct < 16; ++ct) {
            v16bf vf = *(const v16bf*)(bufp + 1024 + (ct * 32 + lane) * 16);
            acc[ct] = __builtin_amdgcn_wmma_f32_16x16x32_bf16(false, vf, false, pf,
                                                              (short)0, acc[ct],
                                                              false, false);
        }

        cur = (cur == 2) ? 0 : cur + 1;
    }

    // epilogue: out = gamma * (acc/l) + x   (coalesced 64B f32 segments)
    const float inv_l = 1.0f / l;
    const float g = gamma[0];
    const int n = it * 16 + lm;
    #pragma unroll
    for (int ct = 0; ct < 16; ++ct) {
        #pragma unroll
        for (int r = 0; r < 8; ++r) {
            int c = ct * 16 + r + half * 8;
            size_t idx = ((size_t)b * C_DIM + c) * N_DIM + n;
            out[idx] = fmaf(g, acc[ct][r] * inv_l, x[idx]);
        }
    }
}

// ---------------------------------------------------------------------------
extern "C" void kernel_launch(void* const* d_in, const int* in_sizes, int n_in,
                              void* d_out, int out_size, void* d_ws, size_t ws_size,
                              hipStream_t stream) {
    const float* x     = (const float*)d_in[0];
    const float* Wq    = (const float*)d_in[1];
    const float* bq    = (const float*)d_in[2];
    const float* Wk    = (const float*)d_in[3];
    const float* bk    = (const float*)d_in[4];
    const float* Wv    = (const float*)d_in[5];
    const float* bv    = (const float*)d_in[6];
    const float* gamma = (const float*)d_in[7];
    float* out = (float*)d_out;

    // workspace: qB 2MB | kv blocks 18MB  (bf16 fragments)
    bf16_t* qB = (bf16_t*)d_ws;
    bf16_t* kv = (bf16_t*)((char*)d_ws + (2u << 20));

    sa_proj_kernel<<<B_DIM * 128, 256, 0, stream>>>(x, Wq, bq, Wk, bk, Wv, bv,
                                                    qB, kv);
    sa_attn_kernel<<<(B_DIM * 256) / 4, 128, 0, stream>>>(qB, kv, x, gamma, out);
}
